// RNNBase_32143535243284
// MI455X (gfx1250) — matrix operand order repaired and checked by reading, not compiled, not run on hardware
//
#include <hip/hip_runtime.h>

#define B_  64
#define T_  1024
#define I_  256
#define H_  512
#define G4H 2048   // 4*H

typedef __attribute__((ext_vector_type(16))) __bf16        v16bf;
typedef __attribute__((ext_vector_type(8)))  float         v8f;
typedef __attribute__((ext_vector_type(4)))  unsigned int  u32x4;

union Frag16 { v16bf bf; u32x4 q[2]; };

__device__ __forceinline__ unsigned short f2bf(float f) {
  union { float f; unsigned u; } x; x.f = f;
  unsigned r = x.u + 0x7FFFu + ((x.u >> 16) & 1u);   // round-to-nearest-even
  return (unsigned short)(r >> 16);
}

__device__ __forceinline__ float sigm(float x) { return 1.0f / (1.0f + __expf(-x)); }

// A 16x32 bf16 fragment (ISA 7.12.2): lane L -> row m = L&15;
// lanes 0-15 hold K {0..7,16..23}, lanes 16-31 hold K {8..15,24..31}.
__device__ __forceinline__ Frag16 load_a_k(const unsigned short* __restrict__ p, int k0) {
  Frag16 f;
  f.q[0] = *(const u32x4*)(p + k0);
  f.q[1] = *(const u32x4*)(p + k0 + 16);
  return f;
}
// B 32x16 bf16 fragment (B = W^T, W row-major): lane L -> col n = L&15;
// lanes 0-15 hold K {0..7,16..23}, lanes 16-31 hold K {8..15,24..31}.
__device__ __forceinline__ Frag16 load_b_k(const unsigned short* __restrict__ p, int k0) {
  Frag16 f;
  f.q[0] = *(const u32x4*)(p + k0);
  f.q[1] = *(const u32x4*)(p + k0 + 16);
  return f;
}

// Fully-unrolled K chain with two independent accumulator chains so WMMA
// issue is never blocked on a single RAW dependency.
template <int KTOT>
__device__ __forceinline__ v8f gemm_part(const unsigned short* __restrict__ Ab,
                                         const unsigned short* __restrict__ Bb) {
  v8f a0 = (v8f)(0.0f), a1 = (v8f)(0.0f);
  #pragma unroll
  for (int k0 = 0; k0 < KTOT; k0 += 64) {
    Frag16 x0 = load_a_k(Ab, k0);
    Frag16 w0 = load_b_k(Bb, k0);
    Frag16 x1 = load_a_k(Ab, k0 + 32);
    Frag16 w1 = load_b_k(Bb, k0 + 32);
    a0 = __builtin_amdgcn_wmma_f32_16x16x32_bf16(false, x0.bf, false, w0.bf,
                                                 (short)0, a0, false, false);
    a1 = __builtin_amdgcn_wmma_f32_16x16x32_bf16(false, x1.bf, false, w1.bf,
                                                 (short)0, a1, false, false);
  }
  return a0 + a1;
}

struct StepArgs {
  const unsigned short* Xin;   long ldx;               // input activations [64, KIN]
  const unsigned short* Wih;                           // [2048, KIN] bf16 row-major
  const unsigned short* Whh;                           // [2048, 512] bf16 row-major
  const float*          bsum;                          // [2048] (b_ih + b_hh)
  const unsigned short* Hprev;                         // [64, 512] bf16
  float*                Cst;                           // [64, 512] fp32 (in/out)
  unsigned short*       Hnext;                         // [64, 512] bf16
  float* OutH;   long outStrideB;                      // optional fp32 h output
  float* FinalH; float* FinalC;                        // optional final-state dump
};

// One LSTM sub-step executed by one block (16 batch rows x 16 hidden cols,
// all 4 gates). 8 waves: wave w -> gate g = w&3, part = w>>2
// (part 0: Xin*Wih^T with K=KIN; part 1: Hprev*Whh^T with K=512).
// Both the part select and the trip counts are wave-uniform/compile-time.
template <int KIN>
__device__ __forceinline__ void step_body(const StepArgs& A, int blk, float* lds) {
  const int row0  = (blk >> 5) << 4;          // batch tile base (4 M tiles)
  const int nbase = (blk & 31) << 4;          // hidden column base (32 N tiles)
  const int wv    = threadIdx.x >> 5;
  const int lane  = threadIdx.x & 31;
  const int g     = wv & 3;

  const int aRow = row0 + (lane & 15);
  const int aOff = (lane >> 4) << 3;
  const int bRow = g * H_ + nbase + (lane & 15);
  const int bOff = (lane >> 4) << 3;

  v8f acc;
  if (wv < 4) {   // input product, K = KIN (wave-uniform branch)
    const unsigned short* Ab = A.Xin + (long)aRow * A.ldx + aOff;
    const unsigned short* Bb = A.Wih + (long)bRow * KIN + bOff;
    acc = gemm_part<KIN>(Ab, Bb);
  } else {        // recurrent product, K = 512
    const unsigned short* Ab = A.Hprev + (long)aRow * H_ + aOff;
    const unsigned short* Bb = A.Whh + (long)bRow * H_ + bOff;
    acc = gemm_part<H_>(Ab, Bb);
  }

  // C layout: element r, lane L  <->  (m = r + 8*(L>>4), n = L&15)
  {
    float* dst = lds + wv * 256 + (lane & 15);
    const int mhi = (lane >> 4) << 3;
    #pragma unroll
    for (int r = 0; r < 8; ++r) dst[(r + mhi) * 16] = acc[r];
  }
  __syncthreads();

  // epilogue: one (b,j) element per thread
  const int e = threadIdx.x;                  // 0..255
  const int b = row0 + (e >> 4);
  const int j = nbase + (e & 15);
  const float gi = sigm (lds[0 * 256 + e] + lds[4 * 256 + e] + A.bsum[j]);
  const float gf = sigm (lds[1 * 256 + e] + lds[5 * 256 + e] + A.bsum[H_ + j]);
  const float gg = tanhf(lds[2 * 256 + e] + lds[6 * 256 + e] + A.bsum[2 * H_ + j]);
  const float go = sigm (lds[3 * 256 + e] + lds[7 * 256 + e] + A.bsum[3 * H_ + j]);
  const long idx = (long)b * H_ + j;
  const float c = gf * A.Cst[idx] + gi * gg;
  const float h = go * tanhf(c);
  A.Cst[idx]   = c;
  A.Hnext[idx] = f2bf(h);
  if (A.OutH)   A.OutH[(long)b * A.outStrideB + j] = h;
  if (A.FinalH) { A.FinalH[idx] = h; A.FinalC[idx] = c; }
}

// Pipeline start: layer0 step 0 only.
__global__ __launch_bounds__(256)
void first_step_kernel(StepArgs a0) {
  __shared__ float lds[8 * 256];
  step_body<I_>(a0, blockIdx.x, lds);
}
// Steady state: blocks [0,128) = layer0 @ t, blocks [128,256) = layer1 @ t-1.
__global__ __launch_bounds__(256)
void fused_step_kernel(StepArgs a0, StepArgs a1) {
  __shared__ float lds[8 * 256];
  const int blk = blockIdx.x;
  if (blk < 128) step_body<I_>(a0, blk, lds);
  else           step_body<H_>(a1, blk - 128, lds);
}
// Pipeline drain: layer1 final step only.
__global__ __launch_bounds__(256)
void last_step_kernel(StepArgs a1) {
  __shared__ float lds[8 * 256];
  step_body<H_>(a1, blockIdx.x, lds);
}

// ---- prep kernels -----------------------------------------------------------
__global__ void cvt_bf16_kernel(const float* __restrict__ in,
                                unsigned short* __restrict__ out, long n) {
  long i = (long)blockIdx.x * blockDim.x + threadIdx.x;
  if (i < n) out[i] = f2bf(in[i]);
}
__global__ void bias_sum_kernel(const float* __restrict__ a,
                                const float* __restrict__ b,
                                float* __restrict__ out, int n) {
  int i = blockIdx.x * blockDim.x + threadIdx.x;
  if (i < n) out[i] = a[i] + b[i];
}
__global__ void zero_f32_kernel(float* __restrict__ p, long n) {
  long i = (long)blockIdx.x * blockDim.x + threadIdx.x;
  if (i < n) p[i] = 0.0f;
}
__global__ void zero_u16_kernel(unsigned short* __restrict__ p, long n) {
  long i = (long)blockIdx.x * blockDim.x + threadIdx.x;
  if (i < n) p[i] = 0;
}

extern "C" void kernel_launch(void* const* d_in, const int* in_sizes, int n_in,
                              void* d_out, int out_size, void* d_ws, size_t ws_size,
                              hipStream_t stream) {
  const float* x    = (const float*)d_in[0];
  const float* Wih0 = (const float*)d_in[1];
  const float* Whh0 = (const float*)d_in[2];
  const float* bih0 = (const float*)d_in[3];
  const float* bhh0 = (const float*)d_in[4];
  const float* Wih1 = (const float*)d_in[5];
  const float* Whh1 = (const float*)d_in[6];
  const float* bih1 = (const float*)d_in[7];
  const float* bhh1 = (const float*)d_in[8];
  float* out = (float*)d_out;

  // ---- workspace layout ----
  char* ws = (char*)d_ws;
  size_t off = 0;
  auto take = [&](size_t bytes) { char* p = ws + off; off += (bytes + 255) & ~(size_t)255; return p; };
  unsigned short* xbf   = (unsigned short*)take((size_t)B_ * T_ * I_ * 2);
  unsigned short* Wih0b = (unsigned short*)take((size_t)G4H * I_ * 2);
  unsigned short* Whh0b = (unsigned short*)take((size_t)G4H * H_ * 2);
  unsigned short* Wih1b = (unsigned short*)take((size_t)G4H * H_ * 2);
  unsigned short* Whh1b = (unsigned short*)take((size_t)G4H * H_ * 2);
  float*          bsum0 = (float*)take((size_t)G4H * 4);
  float*          bsum1 = (float*)take((size_t)G4H * 4);
  float*          c0    = (float*)take((size_t)B_ * H_ * 4);
  float*          c1    = (float*)take((size_t)B_ * H_ * 4);
  unsigned short* h0bf  = (unsigned short*)take((size_t)2 * B_ * H_ * 2);  // double buffer
  unsigned short* h1bf  = (unsigned short*)take((size_t)2 * B_ * H_ * 2);

  // ---- precision prep ----
  {
    long n = (long)B_ * T_ * I_;
    cvt_bf16_kernel<<<dim3((unsigned)((n + 255) / 256)), dim3(256), 0, stream>>>(x, xbf, n);
  }
  cvt_bf16_kernel<<<dim3((G4H * I_) / 256), dim3(256), 0, stream>>>(Wih0, Wih0b, (long)G4H * I_);
  cvt_bf16_kernel<<<dim3((G4H * H_) / 256), dim3(256), 0, stream>>>(Whh0, Whh0b, (long)G4H * H_);
  cvt_bf16_kernel<<<dim3((G4H * H_) / 256), dim3(256), 0, stream>>>(Wih1, Wih1b, (long)G4H * H_);
  cvt_bf16_kernel<<<dim3((G4H * H_) / 256), dim3(256), 0, stream>>>(Whh1, Whh1b, (long)G4H * H_);
  bias_sum_kernel<<<dim3(G4H / 256), dim3(256), 0, stream>>>(bih0, bhh0, bsum0, G4H);
  bias_sum_kernel<<<dim3(G4H / 256), dim3(256), 0, stream>>>(bih1, bhh1, bsum1, G4H);

  // ---- zero state every call (graph replay determinism) ----
  zero_f32_kernel<<<dim3((B_ * H_) / 256), dim3(256), 0, stream>>>(c0, (long)B_ * H_);
  zero_f32_kernel<<<dim3((B_ * H_) / 256), dim3(256), 0, stream>>>(c1, (long)B_ * H_);
  zero_u16_kernel<<<dim3((2 * B_ * H_) / 256), dim3(256), 0, stream>>>(h0bf, (long)2 * B_ * H_);
  zero_u16_kernel<<<dim3((2 * B_ * H_) / 256), dim3(256), 0, stream>>>(h1bf, (long)2 * B_ * H_);

  const long bh = (long)B_ * H_;

  auto mkL0 = [&](int t) {
    StepArgs a{};
    a.Xin  = xbf + (size_t)t * I_;  a.ldx = (long)T_ * I_;
    a.Wih  = Wih0b;  a.Whh = Whh0b;  a.bsum = bsum0;
    a.Hprev = h0bf + (size_t)((t & 1) ^ 1) * bh;
    a.Cst   = c0;
    a.Hnext = h0bf + (size_t)(t & 1) * bh;
    a.OutH = nullptr; a.outStrideB = 0; a.FinalH = nullptr; a.FinalC = nullptr;
    return a;
  };
  auto mkL1 = [&](int s) {
    StepArgs a{};
    a.Xin  = h0bf + (size_t)(s & 1) * bh;  a.ldx = (long)H_;
    a.Wih  = Wih1b;  a.Whh = Whh1b;  a.bsum = bsum1;
    a.Hprev = h1bf + (size_t)((s & 1) ^ 1) * bh;
    a.Cst   = c1;
    a.Hnext = h1bf + (size_t)(s & 1) * bh;
    a.OutH  = out + (size_t)s * H_;  a.outStrideB = (long)T_ * H_;
    const bool last = (s == T_ - 1);
    a.FinalH = last ? (out + (size_t)B_ * T_ * H_)      : nullptr;
    a.FinalC = last ? (out + (size_t)B_ * T_ * H_ + bh) : nullptr;
    return a;
  };

  // pipelined schedule: launch k computes {L0(t=k), L1(t=k-1)}; depth = T+1
  first_step_kernel<<<dim3(128), dim3(256), 0, stream>>>(mkL0(0));
  for (int t = 1; t < T_; ++t) {
    fused_step_kernel<<<dim3(256), dim3(256), 0, stream>>>(mkL0(t), mkL1(t - 1));
  }
  last_step_kernel<<<dim3(128), dim3(256), 0, stream>>>(mkL1(T_ - 1));
}
